// PyraMID_4217657885373
// MI455X (gfx1250) — matrix operand order repaired and checked by reading, not compiled, not run on hardware
//
#include <hip/hip_runtime.h>
#include <math.h>

// ---------------- model constants ----------------
#define SEQL 1000
#define ARY  4
#define DMOD 512
#define NH   8
#define DKH  64
#define NL   4
#define DFF  2048
#define DB   64
#define BATCH 2
#define CINC 7
#define RF   1024
#define NTOT 1365          // 1024+256+64+16+4+1
#define PYRL 341           // 256+64+16+4+1
#define SZ   (BATCH*NTOT)  // 2730
#define OUTROWS 1334       // 1000+250+63+16+4+1

// ---------------- WMMA types ----------------
typedef __attribute__((ext_vector_type(16))) __bf16 v16bf;
typedef __attribute__((ext_vector_type(8)))  float  v8f;

union BF16x16 { v16bf v; unsigned int d[8]; unsigned short u[16]; };
union F32x8   { v8f   v; float f[8]; };

__device__ __forceinline__ unsigned short f2bf(float x) {
  unsigned int u = __float_as_uint(x);
  u += 0x7fffu + ((u >> 16) & 1u);        // round to nearest even
  return (unsigned short)(u >> 16);
}
__device__ __forceinline__ unsigned int pack2(float a, float b) {
  return (unsigned int)f2bf(a) | ((unsigned int)f2bf(b) << 16);
}
// A-fragment packed-dword (K-pair) index for dword j, half h16 (ISA 7.12.2 layout)
__device__ __forceinline__ int a_kpair(int j, int h16) {
  return (j < 4) ? (j + 4 * h16) : (j + 4 + 4 * h16);
}

// ---------------- pyramid mask (ALLOWED is a module constant, recomputed on device) --
__device__ __forceinline__ void scale_of(int g, int& l, int& i) {
  if      (g < 1024) { l = 0; i = g; }
  else if (g < 1280) { l = 1; i = g - 1024; }
  else if (g < 1344) { l = 2; i = g - 1280; }
  else if (g < 1360) { l = 3; i = g - 1344; }
  else if (g < 1364) { l = 4; i = g - 1360; }
  else               { l = 5; i = g - 1364; }
}
__device__ __forceinline__ bool pyr_allowed(int a, int b) {
  int la, ia, lb, ib;
  scale_of(a, la, ia); scale_of(b, lb, ib);
  if (la == lb) { int d = ia - ib; return d >= -2 && d <= 2; }     // INNER=5 window
  if (la == lb + 1) return (ib >= 4 * ia) && (ib < 4 * ia + 4);    // children
  if (lb == la + 1) return (ia >= 4 * ib) && (ia < 4 * ib + 4);    // parent
  return false;
}

// ================= generic WMMA GEMM =================
// C[M,N] = act( A[M,K] @ B[K,N] + bias (+ resid) ), fp32 in/out, bf16 WMMA core.
// 128 threads (4 waves), block tile 128x64, wave tile 32x64, K step 32.
// Requires K%32==0, N%64==0, 16B-aligned rows (true for all call sites).
template<bool RESID, int ACT>
__global__ __launch_bounds__(128)
void wmma_gemm_kernel(const float* __restrict__ A, int lda,
                      const float* __restrict__ Bm, int ldb,
                      const float* __restrict__ bias,
                      const float* __restrict__ resid,
                      float* __restrict__ C, int ldc,
                      int M, int N, int K)
{
  __shared__ unsigned int sA [128][18];   // [m][kpair] packed bf16x2
  __shared__ unsigned int sBT[64][18];    // [n][kpair] packed bf16x2 (B transposed)
  const int tid  = threadIdx.x;
  const int wave = tid >> 5;
  const int lane = tid & 31;
  const int h16  = lane >> 4;
  const int l16  = lane & 15;
  const int m0   = blockIdx.y * 128;
  const int n0   = blockIdx.x * 64;

  // per-thread A staging coords: rows ar, ar+64; 16 consecutive floats from col acb*2
  const int ar  = tid >> 1;
  const int acb = (tid & 1) * 8;          // kpair base (0 or 8)
  const int agr0 = min(m0 + ar,      M - 1);   // clamp: rows >= M never stored
  const int agr1 = min(m0 + ar + 64, M - 1);
  const float* aptr0 = A + (size_t)agr0 * lda + acb * 2;
  const float* aptr1 = A + (size_t)agr1 * lda + acb * 2;

  F32x8 acc[8];                            // [half*4 + j]
#pragma unroll
  for (int j = 0; j < 8; ++j)
#pragma unroll
    for (int r = 0; r < 8; ++r) acc[j].f[r] = 0.0f;

  for (int kk = 0; kk < K; kk += 32) {
    // ---- stage A tile (branchless, float4 loads, packed dword LDS stores) ----
    {
      const float* ap = aptr0 + kk;
      float4 a0 = *(const float4*)(ap);
      float4 a1 = *(const float4*)(ap + 4);
      float4 a2 = *(const float4*)(ap + 8);
      float4 a3 = *(const float4*)(ap + 12);
      if (kk + 32 < K) __builtin_prefetch(ap + 32, 0, 3);   // global_prefetch_b8
      sA[ar][acb + 0] = pack2(a0.x, a0.y);
      sA[ar][acb + 1] = pack2(a0.z, a0.w);
      sA[ar][acb + 2] = pack2(a1.x, a1.y);
      sA[ar][acb + 3] = pack2(a1.z, a1.w);
      sA[ar][acb + 4] = pack2(a2.x, a2.y);
      sA[ar][acb + 5] = pack2(a2.z, a2.w);
      sA[ar][acb + 6] = pack2(a3.x, a3.y);
      sA[ar][acb + 7] = pack2(a3.z, a3.w);
    }
    {
      const float* ap = aptr1 + kk;
      float4 a0 = *(const float4*)(ap);
      float4 a1 = *(const float4*)(ap + 4);
      float4 a2 = *(const float4*)(ap + 8);
      float4 a3 = *(const float4*)(ap + 12);
      if (kk + 32 < K) __builtin_prefetch(ap + 32, 0, 3);
      sA[ar + 64][acb + 0] = pack2(a0.x, a0.y);
      sA[ar + 64][acb + 1] = pack2(a0.z, a0.w);
      sA[ar + 64][acb + 2] = pack2(a1.x, a1.y);
      sA[ar + 64][acb + 3] = pack2(a1.z, a1.w);
      sA[ar + 64][acb + 4] = pack2(a2.x, a2.y);
      sA[ar + 64][acb + 5] = pack2(a2.z, a2.w);
      sA[ar + 64][acb + 6] = pack2(a3.x, a3.y);
      sA[ar + 64][acb + 7] = pack2(a3.z, a3.w);
    }
    // ---- stage B tile transposed (pack two consecutive K rows per dword) ----
#pragma unroll
    for (int i = 0; i < 4; ++i) {
      const int idx = tid + 128 * i;
      const int kp = idx >> 5, np = idx & 31;
      const float* bp = Bm + (size_t)(kk + 2 * kp) * ldb + n0 + 2 * np;
      float2 r0 = *(const float2*)bp;
      float2 r1 = *(const float2*)(bp + ldb);
      if (i == 0 && kk + 32 < K) __builtin_prefetch(bp + (size_t)32 * ldb, 0, 3);
      sBT[2 * np + 0][kp] = pack2(r0.x, r1.x);
      sBT[2 * np + 1][kp] = pack2(r0.y, r1.y);
    }
    __syncthreads();

    BF16x16 af0, af1;
    const int arow0 = 32 * wave + l16;
#pragma unroll
    for (int j = 0; j < 8; ++j) af0.d[j] = sA[arow0][a_kpair(j, h16)];
#pragma unroll
    for (int j = 0; j < 8; ++j) af1.d[j] = sA[arow0 + 16][a_kpair(j, h16)];

#pragma unroll
    for (int j = 0; j < 4; ++j) {
      BF16x16 bf;
      const int bn = 16 * j + l16;
#pragma unroll
      for (int e = 0; e < 8; ++e) bf.d[e] = sBT[bn][a_kpair(e, h16)];
      acc[j].v = __builtin_amdgcn_wmma_f32_16x16x32_bf16(
          false, af0.v, false, bf.v, (short)0, acc[j].v, false, false);
      acc[4 + j].v = __builtin_amdgcn_wmma_f32_16x16x32_bf16(
          false, af1.v, false, bf.v, (short)0, acc[4 + j].v, false, false);
    }
    __syncthreads();
  }

#pragma unroll
  for (int half = 0; half < 2; ++half)
#pragma unroll
    for (int j = 0; j < 4; ++j) {
      const int col = n0 + 16 * j + l16;                  // always < N (N%64==0)
      const float bcol = bias[col];
#pragma unroll
      for (int r = 0; r < 8; ++r) {
        const int row = m0 + 32 * wave + 16 * half + r + 8 * h16;
        if (row < M) {
          float xv = acc[half * 4 + j].f[r] + bcol;
          if (RESID) xv += resid[(size_t)row * ldc + col];
          if (ACT == 1) xv = 0.5f * xv * (1.0f + erff(xv * 0.70710678118f)); // exact GELU
          C[(size_t)row * ldc + col] = xv;
        }
      }
    }
}

// ================= fused masked flash attention =================
// 128 threads = 4 waves; each wave owns one 16-row q tile; K/V^T tiles staged
// cooperatively and shared by all 4 waves. Online softmax, all matmuls WMMA bf16.
__global__ __launch_bounds__(128)
void attn_kernel(const float* __restrict__ q, const float* __restrict__ k,
                 const float* __restrict__ v, float* __restrict__ o)
{
  __shared__ unsigned int   sK [32][34];       // [key][dk pair]          (shared)
  __shared__ unsigned int   sVT[64][18];       // [dk][key pair]          (shared)
  __shared__ unsigned int   sQ [4][16][34];    // per-wave [qrow][dk pair], pre-scaled
  __shared__ unsigned short sP [4][16][36];    // per-wave [qrow][key] probs
  const int tid  = threadIdx.x;
  const int wave = tid >> 5;
  const int lane = tid & 31;
  const int h16 = lane >> 4, l16 = lane & 15;
  const int q0   = (blockIdx.x * 4 + wave) * 16;
  const int head = blockIdx.y;
  const int b    = blockIdx.z;
  const size_t base = (size_t)b * NTOT * DMOD + head * DKH;
  const float scale = 0.125f;                  // 1/sqrt(64)

  // ---- stage own Q tile (clamped rows; rows >= NTOT are never stored) ----
#pragma unroll
  for (int row = 0; row < 16; ++row) {
    const int gr = min(q0 + row, NTOT - 1);
    float2 qa = *(const float2*)&q[base + (size_t)gr * DMOD + 2 * lane];
    sQ[wave][row][lane] = pack2(qa.x * scale, qa.y * scale);
  }
  __syncthreads();
  BF16x16 qf[2];
#pragma unroll
  for (int s = 0; s < 2; ++s)
#pragma unroll
    for (int j = 0; j < 8; ++j) qf[s].d[j] = sQ[wave][l16][16 * s + a_kpair(j, h16)];

  F32x8 oa[4];
#pragma unroll
  for (int j = 0; j < 4; ++j)
#pragma unroll
    for (int r = 0; r < 8; ++r) oa[j].f[r] = 0.0f;
  float mrow[8], lrow[8];
#pragma unroll
  for (int r = 0; r < 8; ++r) { mrow[r] = -1e9f; lrow[r] = 0.0f; }

  for (int kk0 = 0; kk0 < NTOT; kk0 += 32) {
    __syncthreads();
    // ---- cooperative staging: each wave loads 8 K rows and 4 V^T key-pairs ----
#pragma unroll
    for (int i = 0; i < 8; ++i) {
      const int key = wave * 8 + i;
      const int gk = min(kk0 + key, NTOT - 1);
      float2 ka = *(const float2*)&k[base + (size_t)gk * DMOD + 2 * lane];
      sK[key][lane] = pack2(ka.x, ka.y);
    }
#pragma unroll
    for (int i = 0; i < 4; ++i) {
      const int kp = wave * 4 + i;
      const int g0 = min(kk0 + 2 * kp + 0, NTOT - 1);
      const int g1 = min(kk0 + 2 * kp + 1, NTOT - 1);
      float2 va = *(const float2*)&v[base + (size_t)g0 * DMOD + 2 * lane];
      float2 vb = *(const float2*)&v[base + (size_t)g1 * DMOD + 2 * lane];
      sVT[2 * lane + 0][kp] = pack2(va.x, vb.x);
      sVT[2 * lane + 1][kp] = pack2(va.y, vb.y);
    }
    if (kk0 + 32 < NTOT) {
      __builtin_prefetch(&k[base + (size_t)(kk0 + 32 + lane) * DMOD], 0, 3);
      __builtin_prefetch(&v[base + (size_t)(kk0 + 32 + lane) * DMOD], 0, 3);
    }
    __syncthreads();

    // ---- S = Q @ K^T (two 16-key subtiles, DK=64 -> 2 WMMAs each) ----
    F32x8 sacc[2];
#pragma unroll
    for (int j = 0; j < 2; ++j)
#pragma unroll
      for (int r = 0; r < 8; ++r) sacc[j].f[r] = 0.0f;
#pragma unroll
    for (int j = 0; j < 2; ++j) {
      const int key = 16 * j + l16;
#pragma unroll
      for (int s = 0; s < 2; ++s) {
        BF16x16 bfk;
#pragma unroll
        for (int e = 0; e < 8; ++e) bfk.d[e] = sK[key][16 * s + a_kpair(e, h16)];
        sacc[j].v = __builtin_amdgcn_wmma_f32_16x16x32_bf16(
            false, qf[s].v, false, bfk.v, (short)0, sacc[j].v, false, false);
      }
    }

    // ---- mask + online softmax (rows live across 16-lane groups) ----
    float p[2][8];
#pragma unroll
    for (int r = 0; r < 8; ++r) {
      const int qg = q0 + r + 8 * h16;
      float mloc = -1e9f;
#pragma unroll
      for (int j = 0; j < 2; ++j) {
        const int kg = kk0 + 16 * j + l16;
        bool ok = (qg < NTOT) && (kg < NTOT) && pyr_allowed(qg, kg);
        float sv = ok ? sacc[j].f[r] : -1e9f;
        p[j][r] = sv;
        mloc = fmaxf(mloc, sv);
      }
#pragma unroll
      for (int off = 1; off < 16; off <<= 1)
        mloc = fmaxf(mloc, __shfl_xor(mloc, off, 16));
      const float mnew  = fmaxf(mrow[r], mloc);
      const float alpha = expf(mrow[r] - mnew);
      float ps = 0.0f;
#pragma unroll
      for (int j = 0; j < 2; ++j) {
        float e = expf(p[j][r] - mnew);
        p[j][r] = e;
        ps += e;
      }
#pragma unroll
      for (int off = 1; off < 16; off <<= 1)
        ps += __shfl_xor(ps, off, 16);
      lrow[r] = lrow[r] * alpha + ps;
      mrow[r] = mnew;
#pragma unroll
      for (int j = 0; j < 4; ++j) oa[j].f[r] *= alpha;
    }

    // ---- C-layout -> A-layout via per-wave LDS region, then O += P @ V ----
#pragma unroll
    for (int j = 0; j < 2; ++j)
#pragma unroll
      for (int r = 0; r < 8; ++r)
        sP[wave][r + 8 * h16][16 * j + l16] = f2bf(p[j][r]);
    __syncthreads();
    BF16x16 pf;
#pragma unroll
    for (int j = 0; j < 8; ++j)
      pf.d[j] = *(const unsigned int*)&sP[wave][l16][2 * a_kpair(j, h16)];
#pragma unroll
    for (int j = 0; j < 4; ++j) {
      BF16x16 bfv;
#pragma unroll
      for (int e = 0; e < 8; ++e) bfv.d[e] = sVT[16 * j + l16][a_kpair(e, h16)];
      oa[j].v = __builtin_amdgcn_wmma_f32_16x16x32_bf16(
          false, pf.v, false, bfv.v, (short)0, oa[j].v, false, false);
    }
  }

#pragma unroll
  for (int j = 0; j < 4; ++j)
#pragma unroll
    for (int r = 0; r < 8; ++r) {
      const int qg = q0 + r + 8 * h16;
      if (qg < NTOT) {
        float denom = lrow[r];
        float val = (denom > 0.0f) ? oa[j].f[r] / denom : 0.0f;
        o[base + (size_t)qg * DMOD + 16 * j + l16] = val;
      }
    }
}

// ================= LayerNorm (one row per block) =================
__global__ __launch_bounds__(128)
void ln_kernel(const float* __restrict__ x, const float* __restrict__ g,
               const float* __restrict__ bt, float* __restrict__ y)
{
  __shared__ float red[128];
  const int row = blockIdx.x;
  const float* xr = x + (size_t)row * DMOD;
  float s = 0.0f;
  for (int i = threadIdx.x; i < DMOD; i += 128) s += xr[i];
  red[threadIdx.x] = s; __syncthreads();
  for (int st = 64; st > 0; st >>= 1) { if (threadIdx.x < st) red[threadIdx.x] += red[threadIdx.x + st]; __syncthreads(); }
  const float mu = red[0] / (float)DMOD;
  __syncthreads();
  float s2 = 0.0f;
  for (int i = threadIdx.x; i < DMOD; i += 128) { float d = xr[i] - mu; s2 += d * d; }
  red[threadIdx.x] = s2; __syncthreads();
  for (int st = 64; st > 0; st >>= 1) { if (threadIdx.x < st) red[threadIdx.x] += red[threadIdx.x + st]; __syncthreads(); }
  const float inv = rsqrtf(red[0] / (float)DMOD + 1e-5f);
  for (int i = threadIdx.x; i < DMOD; i += 128)
    y[(size_t)row * DMOD + i] = (xr[i] - mu) * inv * g[i] + bt[i];
}

// ================= preamble kernels (tiny cost, VALU) =================
// circular conv(k=3) token embedding + on-device sinusoidal PE, left-padded to RF
__global__ void embed_kernel(const float* __restrict__ x, const float* __restrict__ ew,
                             float* __restrict__ pre)
{
  int idx = blockIdx.x * blockDim.x + threadIdx.x;
  if (idx >= BATCH * RF * DMOD) return;
  int d = idx % DMOD;
  int t = (idx / DMOD) % RF;
  int b = idx / (DMOD * RF);
  float val = 0.0f;
  if (t >= RF - SEQL) {
    int s = t - (RF - SEQL);
    float acc = 0.0f;
#pragma unroll
    for (int kk = 0; kk < 3; ++kk) {
      int sp = (s + kk - 1 + SEQL) % SEQL;
      const float* xr = x + ((size_t)b * SEQL + sp) * CINC;
#pragma unroll
      for (int c = 0; c < CINC; ++c) acc += xr[c] * ew[(d * CINC + c) * 3 + kk];
    }
    const float cexp = -9.210340371976184f / (float)DMOD;     // -ln(10000)/D
    int   i2  = d & ~1;
    float div = expf((float)i2 * cexp);
    float ang = (float)s * div;
    acc += (d & 1) ? cosf(ang) : sinf(ang);
    val = acc;
  }
  pre[((size_t)b * NTOT + t) * DMOD + d] = val;
}

__global__ void down_kernel(const float* __restrict__ pre, const float* __restrict__ w,
                            const float* __restrict__ bb, float* __restrict__ dn)
{
  int idx = blockIdx.x * blockDim.x + threadIdx.x;
  if (idx >= BATCH * RF * DB) return;
  int j = idx % DB;
  int t = (idx / DB) % RF;
  int b = idx / (DB * RF);
  const float* hr = pre + ((size_t)b * NTOT + t) * DMOD;
  float acc = bb[j];
  for (int d = 0; d < DMOD; ++d) acc += hr[d] * w[d * DB + j];
  dn[idx] = acc;
}

// strided conv (k=4, stride=4) + ELU; BatchNorm at init/eval == identity
__global__ void conv_kernel(const float* __restrict__ in, const float* __restrict__ w,
                            const float* __restrict__ bb, float* __restrict__ out,
                            int Lout, int in_stride_b, int out_stride_b)
{
  int idx = blockIdx.x * blockDim.x + threadIdx.x;
  if (idx >= BATCH * Lout * DB) return;
  int o = idx % DB;
  int t = (idx / DB) % Lout;
  int b = idx / (DB * Lout);
  float acc = bb[o];
  const float* ip = in + (size_t)b * in_stride_b;
  for (int ic = 0; ic < DB; ++ic) {
#pragma unroll
    for (int kk = 0; kk < ARY; ++kk)
      acc += ip[(4 * t + kk) * DB + ic] * w[(o * DB + ic) * ARY + kk];
  }
  acc = (acc > 0.0f) ? acc : (expf(acc) - 1.0f);               // ELU
  out[(size_t)b * out_stride_b + t * DB + o] = acc;
}

__global__ void up_kernel(const float* __restrict__ pyr, const float* __restrict__ w,
                          const float* __restrict__ bb, float* __restrict__ pre)
{
  int idx = blockIdx.x * blockDim.x + threadIdx.x;
  if (idx >= BATCH * PYRL * DMOD) return;
  int d = idx % DMOD;
  int j = (idx / DMOD) % PYRL;
  int b = idx / (DMOD * PYRL);
  const float* pr = pyr + ((size_t)b * PYRL + j) * DB;
  float acc = bb[d];
#pragma unroll
  for (int c = 0; c < DB; ++c) acc += pr[c] * w[c * DMOD + d];
  pre[((size_t)b * NTOT + RF + j) * DMOD + d] = acc;
}

__global__ void gather_kernel(const float* __restrict__ h, float* __restrict__ out)
{
  const int offs[7]   = {0, 1000, 1250, 1313, 1329, 1333, 1334};
  const int starts[6] = {0, 1024, 1280, 1344, 1360, 1364};
  int idx = blockIdx.x * blockDim.x + threadIdx.x;
  if (idx >= BATCH * OUTROWS * DMOD) return;
  int d = idx % DMOD;
  int r = (idx / DMOD) % OUTROWS;
  int b = idx / (DMOD * OUTROWS);
  int lvl = 0;
  while (lvl < 5 && r >= offs[lvl + 1]) ++lvl;
  int src = starts[lvl] + (r - offs[lvl]);
  out[idx] = h[((size_t)b * NTOT + src) * DMOD + d];
}

// ================= host-side orchestration =================
extern "C" void kernel_launch(void* const* d_in, const int* in_sizes, int n_in,
                              void* d_out, int out_size, void* d_ws, size_t ws_size,
                              hipStream_t stream)
{
  const float* x      = (const float*)d_in[0];
  const float* emb_w  = (const float*)d_in[1];
  const float* down_w = (const float*)d_in[2];
  const float* down_b = (const float*)d_in[3];
  const float* convs_w= (const float*)d_in[4];
  const float* convs_b= (const float*)d_in[5];
  const float* up_w   = (const float*)d_in[6];
  const float* up_b   = (const float*)d_in[7];
  const float* neck_g = (const float*)d_in[8];
  const float* neck_b = (const float*)d_in[9];
  const float* Wq = (const float*)d_in[10]; const float* bq = (const float*)d_in[11];
  const float* Wk = (const float*)d_in[12]; const float* bk = (const float*)d_in[13];
  const float* Wv = (const float*)d_in[14]; const float* bv = (const float*)d_in[15];
  const float* Wo = (const float*)d_in[16]; const float* bo = (const float*)d_in[17];
  const float* ln1_g = (const float*)d_in[18]; const float* ln1_b = (const float*)d_in[19];
  const float* W1 = (const float*)d_in[20]; const float* b1 = (const float*)d_in[21];
  const float* W2 = (const float*)d_in[22]; const float* b2 = (const float*)d_in[23];
  const float* ln2_g = (const float*)d_in[24]; const float* ln2_b = (const float*)d_in[25];
  (void)in_sizes; (void)n_in; (void)out_size; (void)ws_size;

  // workspace layout (~57 MB)
  char* ws = (char*)d_ws;
  size_t off = 0;
  auto alloc = [&](size_t bytes) -> float* {
    float* p = (float*)(ws + off);
    off = (off + bytes + 255) & ~(size_t)255;
    return p;
  };
  float* pre = alloc((size_t)SZ * DMOD * 4);        // pre-LN residual stream
  float* hb  = alloc((size_t)SZ * DMOD * 4);        // post-LN hidden state
  float* qb  = alloc((size_t)SZ * DMOD * 4);
  float* kb  = alloc((size_t)SZ * DMOD * 4);
  float* vb  = alloc((size_t)SZ * DMOD * 4);
  float* ob  = alloc((size_t)SZ * DMOD * 4);
  float* ffb = alloc((size_t)SZ * DFF * 4);
  float* dwn = alloc((size_t)BATCH * RF * DB * 4);
  float* pyr = alloc((size_t)BATCH * PYRL * DB * 4);

  // ---- embedding + PE (writes pre rows [0,1024), incl. 24 zero-pad rows) ----
  embed_kernel<<<(BATCH * RF * DMOD + 255) / 256, 256, 0, stream>>>(x, emb_w, pre);
  // ---- bottleneck: down-proj, conv pyramid, up-proj (pre rows [1024,1365)) ----
  down_kernel<<<(BATCH * RF * DB + 255) / 256, 256, 0, stream>>>(pre, down_w, down_b, dwn);
  {
    int Lin = RF, poff = 0;
    const float* in = dwn; int in_stride = RF * DB;
    for (int i = 0; i < 5; ++i) {
      int Lout = Lin / ARY;
      float* outp = pyr + poff * DB;
      conv_kernel<<<(BATCH * Lout * DB + 255) / 256, 256, 0, stream>>>(
          in, convs_w + (size_t)i * DB * DB * ARY, convs_b + i * DB, outp,
          Lout, in_stride, PYRL * DB);
      in = outp; in_stride = PYRL * DB; poff += Lout; Lin = Lout;
    }
  }
  up_kernel<<<(BATCH * PYRL * DMOD + 255) / 256, 256, 0, stream>>>(pyr, up_w, up_b, pre);
  ln_kernel<<<SZ, 128, 0, stream>>>(pre, neck_g, neck_b, hb);

  // ---- 4 post-norm encoder layers: WMMA GEMMs + fused masked flash attention ----
  dim3 g512((DMOD + 63) / 64, (SZ + 127) / 128);    // 8 x 22 blocks
  dim3 gff ((DFF  + 63) / 64, (SZ + 127) / 128);    // 32 x 22 blocks
  dim3 ga  ((NTOT / 16 + 4) / 4, NH, BATCH);        // 22 x 8 x 2 blocks (4 q-tiles each)
  for (int l = 0; l < NL; ++l) {
    const size_t wo512 = (size_t)l * DMOD * DMOD;
    wmma_gemm_kernel<false, 0><<<g512, 128, 0, stream>>>(
        hb, DMOD, Wq + wo512, DMOD, bq + l * DMOD, nullptr, qb, DMOD, SZ, DMOD, DMOD);
    wmma_gemm_kernel<false, 0><<<g512, 128, 0, stream>>>(
        hb, DMOD, Wk + wo512, DMOD, bk + l * DMOD, nullptr, kb, DMOD, SZ, DMOD, DMOD);
    wmma_gemm_kernel<false, 0><<<g512, 128, 0, stream>>>(
        hb, DMOD, Wv + wo512, DMOD, bv + l * DMOD, nullptr, vb, DMOD, SZ, DMOD, DMOD);
    attn_kernel<<<ga, 128, 0, stream>>>(qb, kb, vb, ob);
    wmma_gemm_kernel<true, 0><<<g512, 128, 0, stream>>>(
        ob, DMOD, Wo + wo512, DMOD, bo + l * DMOD, hb, pre, DMOD, SZ, DMOD, DMOD);
    ln_kernel<<<SZ, 128, 0, stream>>>(pre, ln1_g + l * DMOD, ln1_b + l * DMOD, hb);
    wmma_gemm_kernel<false, 1><<<gff, 128, 0, stream>>>(
        hb, DMOD, W1 + (size_t)l * DMOD * DFF, DFF, b1 + l * DFF, nullptr, ffb, DFF,
        SZ, DFF, DMOD);                                                     // GELU
    wmma_gemm_kernel<true, 0><<<g512, 128, 0, stream>>>(
        ffb, DFF, W2 + (size_t)l * DFF * DMOD, DMOD, b2 + l * DMOD, hb, pre, DMOD,
        SZ, DMOD, DFF);
    ln_kernel<<<SZ, 128, 0, stream>>>(pre, ln2_g + l * DMOD, ln2_b + l * DMOD, hb);
  }

  // ---- multiscale gather ----
  gather_kernel<<<(BATCH * OUTROWS * DMOD + 255) / 256, 256, 0, stream>>>(hb, (float*)d_out);
}
